// GATEncoder_59536836657627
// MI455X (gfx1250) — compile-verified, matrix operand order checked by reading, not run on hardware
//
#include <hip/hip_runtime.h>

#define DCH 128
#define NGC 100000
#define NDC 20000
#define EGGC 1000000
#define EGDC 500000
#define NEG_SLOPE 0.2f

typedef __attribute__((ext_vector_type(16))) __bf16          v16bf;
typedef __attribute__((ext_vector_type(16))) unsigned short  v16u16;
typedef __attribute__((ext_vector_type(8)))  unsigned short  v8u16;
typedef __attribute__((ext_vector_type(8)))  float           v8f;

// fp32 -> bf16 (RNE) in integer domain.
__device__ __forceinline__ unsigned short f2bf(float x) {
  unsigned u = __float_as_uint(x);
  u += 0x7FFFu + ((u >> 16) & 1u);
  return (unsigned short)(u >> 16);
}

// Elementwise fp32 -> bf16 row-major conversion (done once per layer per X).
__global__ void gat_f32_to_bf16(const float* __restrict__ in,
                                unsigned short* __restrict__ out, int n) {
  const int i = blockIdx.x * blockDim.x + threadIdx.x;
  if (i < n) out[i] = f2bf(in[i]);
}

// Prepack W[128x128] fp32 (row-major, W[k][n]) into WMMA B-fragment order:
//   Wp[((tile*4 + kstep)*32 + lane)*16 + i]
// tile = output column tile (8), lane layout per ISA 7.12.2 (col = lane&15,
// K-half by lane>=16, K pairs 0..7 / 16..23 within the 32-wide kstep).
// One block of 256 threads handles the whole matrix.
__global__ void gat_prepack_w(const float* __restrict__ W,
                              unsigned short* __restrict__ Wp) {
  const int lane = threadIdx.x & 31;
  const int tile = threadIdx.x >> 5;
  const int col  = tile * 16 + (lane & 15);
  const int half = lane >> 4;
  #pragma unroll
  for (int ks = 0; ks < 4; ++ks) {
    const int g0 = ks * 32 + half * 8;
    const int g1 = g0 + 16;
    unsigned short* dst = Wp + (((tile * 4 + ks) * 32) + lane) * 16;
    #pragma unroll
    for (int i = 0; i < 8; ++i) {
      dst[i]     = f2bf(W[(g0 + i) * DCH + col]);
      dst[8 + i] = f2bf(W[(g1 + i) * DCH + col]);
    }
  }
}

// ---------------------------------------------------------------------------
// GEMM: Y[N x 128] = Xbf[N x 128](bf16) @ Wp(prepacked bf16), f32 accumulate.
// Block = 256 threads = 8 waves; wave w owns column tile w; blockIdx.x owns a
// 16-row tile. Per K-step: 2x b128 A loads (contiguous), 1x 32B B load
// (contiguous, prepacked), 1x v_wmma_f32_16x16x32_bf16.
// ---------------------------------------------------------------------------
__global__ void __launch_bounds__(256)
gat_gemm128_bf16_wmma(const unsigned short* __restrict__ Xbf,
                      const unsigned short* __restrict__ Wp,
                      float* __restrict__ Y) {
  const int lane = threadIdx.x & 31;
  const int wave = threadIdx.x >> 5;     // 0..7 -> column tile
  const int half = lane >> 4;
  const int m    = lane & 15;
  const size_t rowBase = (size_t)blockIdx.x * 16;
  const int col = wave * 16 + m;
  const unsigned short* xrow  = Xbf + (rowBase + m) * DCH;
  const unsigned short* wlane = Wp + ((size_t)(wave * 4) * 32 + lane) * 16;

  v8f c = {};
  #pragma unroll
  for (int ks = 0; ks < 4; ++ks) {
    const int g0 = ks * 32 + half * 8;
    v8u16 alo = *(const v8u16*)(xrow + g0);
    v8u16 ahi = *(const v8u16*)(xrow + g0 + 16);
    v16u16 ab = __builtin_shufflevector(alo, ahi,
        0, 1, 2, 3, 4, 5, 6, 7, 8, 9, 10, 11, 12, 13, 14, 15);
    v16u16 bb = *(const v16u16*)(wlane + (size_t)ks * 32 * 16);
    v16bf a = __builtin_bit_cast(v16bf, ab);
    v16bf b = __builtin_bit_cast(v16bf, bb);
    c = __builtin_amdgcn_wmma_f32_16x16x32_bf16(
        /*neg_a=*/false, a, /*neg_b=*/false, b,
        /*c_mod=*/(short)0, c, /*reuse_a=*/false, /*reuse_b=*/false);
  }
  #pragma unroll
  for (int r = 0; r < 8; ++r)
    Y[(rowBase + (size_t)(r + half * 8)) * DCH + col] = c[r];
}

// zero accumulators, m = -inf, s = 0
__global__ void gat_init_node(float* __restrict__ mbuf, float* __restrict__ sbuf,
                              float* __restrict__ acc, int N) {
  const int i = blockIdx.x * blockDim.x + threadIdx.x;
  if (i < N * DCH) acc[i] = 0.f;
  if (i < N) { mbuf[i] = -__builtin_inff(); sbuf[i] = 0.f; }
}

// Pass A: score[e] = att . leaky_relu(xl[src] + xr[dst]); segment max via
// ordered-int atomic trick. One wave32 per edge, 4 channels per lane.
__global__ void __launch_bounds__(256)
gat_edge_score(const float* __restrict__ xl, const float* __restrict__ xr,
               const int* __restrict__ src, const int* __restrict__ dst,
               const float* __restrict__ att, float* __restrict__ score,
               float* __restrict__ mbuf, int E) {
  const int e = blockIdx.x * 8 + (threadIdx.x >> 5);
  if (e >= E) return;
  const int lane = threadIdx.x & 31;
  const int s = src[e], d = dst[e];
  const float* pl = xl + (size_t)s * DCH;
  const float* pr = xr + (size_t)d * DCH;
  float acc = 0.f;
  #pragma unroll
  for (int j = 0; j < 4; ++j) {
    const int ch = lane + 32 * j;
    float v = pl[ch] + pr[ch];
    v = (v > 0.f) ? v : NEG_SLOPE * v;
    acc += v * att[ch];
  }
  #pragma unroll
  for (int o = 16; o > 0; o >>= 1) acc += __shfl_xor(acc, o, 32);
  if (lane == 0) {
    score[e] = acc;
    if (acc >= 0.f) atomicMax((int*)(mbuf + d), __float_as_int(acc));
    else            atomicMin((unsigned int*)(mbuf + d),
                              (unsigned int)__float_as_int(acc));
  }
}

// Pass B (fused): a = exp(score - m[dst]); s[dst] += a; acc[dst] += a*xl[src].
__global__ void __launch_bounds__(256)
gat_edge_expaccum(const float* __restrict__ xl,
                  const int* __restrict__ src, const int* __restrict__ dst,
                  const float* __restrict__ score, const float* __restrict__ mbuf,
                  float* __restrict__ sbuf, float* __restrict__ acc, int E) {
  const int e = blockIdx.x * 8 + (threadIdx.x >> 5);
  if (e >= E) return;
  const int lane = threadIdx.x & 31;
  const int s = src[e], d = dst[e];
  const float a = __expf(score[e] - mbuf[d]);   // uniform across wave
  if (lane == 0) atomicAdd(sbuf + d, a);
  const float* pl = xl + (size_t)s * DCH;
  float* pa = acc + (size_t)d * DCH;
  #pragma unroll
  for (int j = 0; j < 4; ++j) {
    const int ch = lane + 32 * j;
    atomicAdd(pa + ch, a * pl[ch]);
  }
}

// Pass C: out = acc / s + bias  (nodes with no in-edges -> bias only)
__global__ void gat_finalize(const float* __restrict__ acc, const float* __restrict__ sbuf,
                             const float* __restrict__ bias, float* __restrict__ out, int N) {
  const int i = blockIdx.x * blockDim.x + threadIdx.x;
  if (i >= N * DCH) return;
  const int n = i >> 7, ch = i & (DCH - 1);
  const float sv = sbuf[n];
  const float v = (sv > 0.f) ? acc[i] / sv : 0.f;
  out[i] = v + bias[ch];
}

extern "C" void kernel_launch(void* const* d_in, const int* in_sizes, int n_in,
                              void* d_out, int out_size, void* d_ws, size_t ws_size,
                              hipStream_t stream) {
  (void)in_sizes; (void)n_in; (void)out_size; (void)ws_size;
  const float* x_gene  = (const float*)d_in[0];
  const float* x_dis   = (const float*)d_in[1];
  const int*   src_gg  = (const int*)d_in[2];
  const int*   dst_gg  = (const int*)d_in[3];
  const int*   src_gd  = (const int*)d_in[4];
  const int*   dst_gd  = (const int*)d_in[5];
  const float* Wsrc_gg = (const float*)d_in[6];
  const float* Wdst_gg = (const float*)d_in[7];
  const float* att_gg  = (const float*)d_in[8];
  const float* b_gg    = (const float*)d_in[9];
  const float* Wsrc_gd = (const float*)d_in[10];
  const float* Wdst_gd = (const float*)d_in[11];
  const float* att_gd  = (const float*)d_in[12];
  const float* b_gd    = (const float*)d_in[13];

  float* ws = (float*)d_ws;
  size_t off = 0;
  auto carve = [&](size_t nFloats) { float* p = ws + off; off += nFloats; return p; };
  float* xl     = carve((size_t)NGC * DCH);      // source-side transform (reused gg->gd)
  float* xr     = carve((size_t)NGC * DCH);      // target-side transform (reused gg->gd)
  float* acc    = carve((size_t)NGC * DCH);      // unnormalized numerator (reused g->d)
  float* xg_buf = carve((size_t)NGC * DCH);      // layer-0 gene output
  float* xd_buf = carve((size_t)NDC * DCH);      // layer-0 disease output
  float* score  = carve((size_t)EGGC);           // per-edge scores (gg >= gd)
  float* mbuf   = carve((size_t)NGC);
  float* sbuf   = carve((size_t)NGC);
  unsigned short* xg_bf = (unsigned short*)carve((size_t)NGC * DCH / 2);  // bf16 Xg
  unsigned short* xd_bf = (unsigned short*)carve((size_t)NDC * DCH / 2);  // bf16 Xd
  unsigned short* wp    = (unsigned short*)carve(8192);                   // prepacked W

  float* out_g = (float*)d_out;
  float* out_d = (float*)d_out + (size_t)NGC * DCH;

  for (int l = 0; l < 2; ++l) {
    const float* Xg = (l == 0) ? x_gene : xg_buf;
    const float* Xd = (l == 0) ? x_dis  : xd_buf;
    float* outG = (l == 0) ? xg_buf : out_g;
    float* outD = (l == 0) ? xd_buf : out_d;
    const int wofs = l * DCH * DCH, vofs = l * DCH;

    // Convert this layer's node features to bf16 once (Xg feeds 3 GEMMs).
    gat_f32_to_bf16<<<(NGC * DCH + 255) / 256, 256, 0, stream>>>(Xg, xg_bf, NGC * DCH);
    gat_f32_to_bf16<<<(NDC * DCH + 255) / 256, 256, 0, stream>>>(Xd, xd_bf, NDC * DCH);

    // ---- gene -> gene ----
    gat_prepack_w<<<1, 256, 0, stream>>>(Wsrc_gg + wofs, wp);
    gat_gemm128_bf16_wmma<<<NGC / 16, 256, 0, stream>>>(xg_bf, wp, xl);
    gat_prepack_w<<<1, 256, 0, stream>>>(Wdst_gg + wofs, wp);
    gat_gemm128_bf16_wmma<<<NGC / 16, 256, 0, stream>>>(xg_bf, wp, xr);
    gat_init_node<<<(NGC * DCH + 255) / 256, 256, 0, stream>>>(mbuf, sbuf, acc, NGC);
    gat_edge_score<<<(EGGC + 7) / 8, 256, 0, stream>>>(xl, xr, src_gg, dst_gg,
                                                       att_gg + vofs, score, mbuf, EGGC);
    gat_edge_expaccum<<<(EGGC + 7) / 8, 256, 0, stream>>>(xl, src_gg, dst_gg,
                                                          score, mbuf, sbuf, acc, EGGC);
    gat_finalize<<<(NGC * DCH + 255) / 256, 256, 0, stream>>>(acc, sbuf, b_gg + vofs, outG, NGC);

    // ---- gene -> disease (reads OLD Xg; buffers reused) ----
    gat_prepack_w<<<1, 256, 0, stream>>>(Wsrc_gd + wofs, wp);
    gat_gemm128_bf16_wmma<<<NGC / 16, 256, 0, stream>>>(xg_bf, wp, xl);
    gat_prepack_w<<<1, 256, 0, stream>>>(Wdst_gd + wofs, wp);
    gat_gemm128_bf16_wmma<<<NDC / 16, 256, 0, stream>>>(xd_bf, wp, xr);
    gat_init_node<<<(NDC * DCH + 255) / 256, 256, 0, stream>>>(mbuf, sbuf, acc, NDC);
    gat_edge_score<<<(EGDC + 7) / 8, 256, 0, stream>>>(xl, xr, src_gd, dst_gd,
                                                       att_gd + vofs, score, mbuf, EGDC);
    gat_edge_expaccum<<<(EGDC + 7) / 8, 256, 0, stream>>>(xl, src_gd, dst_gd,
                                                          score, mbuf, sbuf, acc, EGDC);
    gat_finalize<<<(NDC * DCH + 255) / 256, 256, 0, stream>>>(acc, sbuf, b_gd + vofs, outD, NDC);
  }
}
